// PointNetPlusPlus_3599182594097
// MI455X (gfx1250) — compile-verified, matrix operand order checked by reading, not run on hardware
//
#include <hip/hip_runtime.h>
#include <hip/hip_bf16.h>
#include <math.h>

typedef __attribute__((ext_vector_type(16))) _Float16 v16h;
typedef __attribute__((ext_vector_type(8)))  _Float16 v8h;
typedef __attribute__((ext_vector_type(8)))  float    v8f;

#define BNS 0.9999950000374997f   // 1/sqrt(1+1e-5)

// ---------------------------------------------------------------- transpose
__global__ void transpose_xy(const float* __restrict__ xy, float* __restrict__ xyz,
                             int B, int N) {
    int i = blockIdx.x * blockDim.x + threadIdx.x;
    if (i < B * N) {
        int b = i / N, n = i % N;
        xyz[(size_t)i * 2 + 0] = xy[(size_t)b * 2 * N + n];
        xyz[(size_t)i * 2 + 1] = xy[(size_t)b * 2 * N + N + n];
    }
}

// ---------------------------------------------------------------- FPS (one WG per batch)
__global__ void fps_kernel(const float* __restrict__ xyz, int n, int npoint,
                           int* __restrict__ idxout, float* __restrict__ newxyz) {
    __shared__ float px[4096], py[4096], dist[4096];
    __shared__ float rv[256];
    __shared__ int   ri[256];
    __shared__ int   s_far;
    int b = blockIdx.x, tid = threadIdx.x;
    for (int i = tid; i < n; i += blockDim.x) {
        px[i] = xyz[((size_t)b * n + i) * 2 + 0];
        py[i] = xyz[((size_t)b * n + i) * 2 + 1];
        dist[i] = 1e10f;
    }
    if (tid == 0) s_far = 0;
    __syncthreads();
    for (int it = 0; it < npoint; ++it) {
        int far = s_far;
        if (tid == 0) {
            idxout[(size_t)b * npoint + it] = far;
            newxyz[((size_t)b * npoint + it) * 2 + 0] = px[far];
            newxyz[((size_t)b * npoint + it) * 2 + 1] = py[far];
        }
        float fx = px[far], fy = py[far];
        float bv = -1.f; int bi = 0;
        for (int i = tid; i < n; i += blockDim.x) {
            float dx = px[i] - fx, dy = py[i] - fy;
            float nd = fminf(dist[i], dx * dx + dy * dy);
            dist[i] = nd;
            if (nd > bv) { bv = nd; bi = i; }
        }
        rv[tid] = bv; ri[tid] = bi;
        __syncthreads();
        for (int s = 128; s > 0; s >>= 1) {
            if (tid < s) {
                if (rv[tid + s] > rv[tid] ||
                    (rv[tid + s] == rv[tid] && ri[tid + s] < ri[tid])) {
                    rv[tid] = rv[tid + s]; ri[tid] = ri[tid + s];
                }
            }
            __syncthreads();
        }
        if (tid == 0) s_far = ri[0];
        __syncthreads();
    }
}

// ---------------------------------------------------------------- ball query (1 wave / center)
__global__ void ballquery_kernel(const float* __restrict__ xyz,
                                 const float* __restrict__ centers,
                                 int n, int npoint, float r2, int k,
                                 int* __restrict__ out) {
    int gw   = (blockIdx.x * blockDim.x + threadIdx.x) >> 5;
    int lane = threadIdx.x & 31;
    int b = gw / npoint;
    float cx = centers[(size_t)gw * 2 + 0], cy = centers[(size_t)gw * 2 + 1];
    const float* px = xyz + (size_t)b * n * 2;
    int count = 0, first = -1;
    for (int base = 0; base < n && count < k; base += 32) {
        int i = base + lane;
        bool in = false;
        if (i < n) {
            float dx = px[(size_t)i * 2] - cx, dy = px[(size_t)i * 2 + 1] - cy;
            in = (dx * dx + dy * dy) <= r2;
        }
        unsigned int m = (unsigned int)__ballot(in);
        if (in) {
            int pos = count + __popc(m & ((1u << lane) - 1u));
            if (pos < k) out[(size_t)gw * k + pos] = i;
        }
        if (first < 0 && m) first = base + __ffs(m) - 1;
        count += __popc(m);
    }
    if (first < 0) first = 0;
    int start = count < k ? count : k;
    for (int j = start + lane; j < k; j += 32) out[(size_t)gw * k + j] = first;
}

// ---------------------------------------------------------------- weight prep
// Re-swizzle W (K x C, f32) into the per-lane WMMA B-fragment layout (f16, K padded
// to mult. of 32) and fold BN: s = BNS*gamma, t = beta + bias*s.
__global__ void prep_weights(const float* __restrict__ W, const float* __restrict__ b,
                             const float* __restrict__ g, const float* __restrict__ be,
                             int K, int Kpad, int C,
                             _Float16* __restrict__ wf, float* __restrict__ sb) {
    int tid = blockIdx.x * blockDim.x + threadIdx.x;
    int total = Kpad * C;
    if (tid < total) {
        int j    = tid & 15;
        int lane = (tid >> 4) & 31;
        int t2   = tid >> 9;
        int NT   = C >> 4;
        int nt   = t2 % NT;
        int kt   = t2 / NT;
        int col  = nt * 16 + (lane & 15);
        int kb   = (lane >= 16) ? 8 : 0;
        int k    = (j < 8) ? (kt * 32 + kb + j) : (kt * 32 + 16 + kb + (j - 8));
        float v  = (k < K) ? W[(size_t)k * C + col] : 0.f;
        wf[tid] = (_Float16)v;
    }
    if (tid < C) {
        float s = BNS * g[tid];
        sb[tid]     = s;
        sb[C + tid] = be[tid] + b[tid] * s;
    }
}

// ---------------------------------------------------------------- fused MLP layer via WMMA
__device__ __forceinline__ void atomicMaxPosF(float* a, float v) {
    atomicMax((int*)a, __float_as_int(v));
}

template <int KT, int NT, bool FINAL>
__device__ __forceinline__ void mlp_layer(const _Float16* __restrict__ in, int sin,
                                          const _Float16* __restrict__ w,
                                          const float* __restrict__ sb,
                                          _Float16* __restrict__ out, int sout,
                                          int lane, float* __restrict__ lmax) {
    const int ln15  = lane & 15;
    const int kb    = (lane >> 4) << 3;          // 0 or 8
    const int rbase = (lane >> 4) << 3;          // rows 0-7 or 8-15
    for (int nt = 0; nt < NT; ++nt) {
        v8f acc = {0.f, 0.f, 0.f, 0.f, 0.f, 0.f, 0.f, 0.f};
#pragma unroll
        for (int kt = 0; kt < KT; ++kt) {
            const _Float16* p = in + ln15 * sin + kt * 32;
            v8h a0 = *(const v8h*)(p + kb);
            v8h a1 = *(const v8h*)(p + 16 + kb);
            v16h a = __builtin_shufflevector(a0, a1, 0, 1, 2, 3, 4, 5, 6, 7,
                                             8, 9, 10, 11, 12, 13, 14, 15);
            v16h bm = *(const v16h*)(w + ((size_t)(kt * NT + nt) * 32 + lane) * 16);
            acc = __builtin_amdgcn_wmma_f32_16x16x32_f16(false, a, false, bm,
                                                         (short)0, acc, false, false);
        }
        int col = nt * 16 + ln15;
        float s = sb[col], t = sb[NT * 16 + col];
#pragma unroll
        for (int r = 0; r < 8; ++r) {
            float v = fmaxf(acc[r] * s + t, 0.f);
            if (FINAL) atomicMaxPosF(lmax + col, v);
            else       out[(rbase + r) * sout + col] = (_Float16)v;
        }
    }
    __builtin_amdgcn_wave_barrier();  // LDS is in-order per wave; block sched reordering
}

// Fused gather -> MLP(3 layers, WMMA) -> maxpool.
// MODE 0: g = rel_xyz(2).  MODE 1: g = [pts(CP), rel_xyz(2)].  MODE 2: g = [xyz(2), pts(CP)], idx = identity.
template <int MODE, int CINP, int CP, int C1, int C2, int C3, int KS, int CPW, bool OUTF32>
__global__ void sa_fused(const float* __restrict__ srcxyz,
                         const _Float16* __restrict__ pts,
                         const float* __restrict__ centers,
                         const int* __restrict__ ballidx,
                         const _Float16* __restrict__ w1, const _Float16* __restrict__ w2,
                         const _Float16* __restrict__ w3,
                         const float* __restrict__ sb1, const float* __restrict__ sb2,
                         const float* __restrict__ sb3,
                         void* __restrict__ outp,
                         int npoint, int nsrc, int out_stride, int out_coloff) {
    constexpr int S_in  = CINP > C1 ? (CINP > C2 ? CINP : C2) : (C1 > C2 ? C1 : C2);
    constexpr int S_out = C1 > C2 ? C1 : C2;
    constexpr int NTILES = KS / 16;

    extern __shared__ __align__(32) char smem[];
    const int nw   = blockDim.x >> 5;
    const int wave = threadIdx.x >> 5;
    const int lane = threadIdx.x & 31;
    _Float16* inb  = (_Float16*)smem + (size_t)wave * 16 * (S_in + S_out);
    _Float16* outb = inb + 16 * S_in;
    float* lmax = (float*)((_Float16*)smem + (size_t)nw * 16 * (S_in + S_out));

    for (int i = threadIdx.x; i < CPW * C3; i += blockDim.x) lmax[i] = 0.f;
    __syncthreads();

    const int wgbase = blockIdx.x * CPW;
    for (int t = wave; t < CPW * NTILES; t += nw) {
        int cl = t / NTILES, tc = t % NTILES;
        int gc = wgbase + cl;
        int b  = gc / npoint;
        float cx = 0.f, cy = 0.f;
        if (MODE != 2) { cx = centers[(size_t)gc * 2]; cy = centers[(size_t)gc * 2 + 1]; }
        // gather 16 rows into LDS (f16, zero-padded to CINP)
        for (int r = 0; r < 16; ++r) {
            int j = tc * 16 + r;
            int pidx = (MODE == 2) ? j : ballidx[(size_t)gc * KS + j];
            size_t gp = (size_t)b * nsrc + pidx;
            float px = srcxyz[gp * 2], py = srcxyz[gp * 2 + 1];
            _Float16 rx = (_Float16)((MODE == 2) ? px : px - cx);
            _Float16 ry = (_Float16)((MODE == 2) ? py : py - cy);
            for (int ch = lane; ch < CINP; ch += 32) {
                _Float16 v = (_Float16)0.f;
                if (MODE == 0)      { v = (ch == 0) ? rx : (ch == 1) ? ry : (_Float16)0.f; }
                else if (MODE == 1) { v = (ch < CP) ? pts[gp * CP + ch]
                                        : (ch == CP) ? rx : (ch == CP + 1) ? ry : (_Float16)0.f; }
                else                { v = (ch == 0) ? rx : (ch == 1) ? ry
                                        : (ch - 2 < CP) ? pts[gp * CP + (ch - 2)] : (_Float16)0.f; }
                inb[r * S_in + ch] = v;
            }
        }
        __builtin_amdgcn_wave_barrier();
        mlp_layer<CINP / 32, C1 / 16, false>(inb, S_in, w1, sb1, outb, S_out, lane, nullptr);
        mlp_layer<C1 / 32, C2 / 16, false>(outb, S_out, w2, sb2, inb, S_in, lane, nullptr);
        mlp_layer<C2 / 32, C3 / 16, true>(inb, S_in, w3, sb3, nullptr, 0, lane, lmax + cl * C3);
    }
    __syncthreads();
    for (int i = threadIdx.x; i < CPW * C3; i += blockDim.x) {
        int cl = i / C3, c = i % C3;
        size_t o = (size_t)(wgbase + cl) * out_stride + out_coloff + c;
        if (OUTF32) ((float*)outp)[o] = lmax[i];
        else        ((_Float16*)outp)[o] = (_Float16)lmax[i];
    }
}

// ---------------------------------------------------------------- head (tiny)
__global__ void head_kernel(const float* __restrict__ l3,
                            const float* __restrict__ W1, const float* __restrict__ b1,
                            const float* __restrict__ g1, const float* __restrict__ be1,
                            const float* __restrict__ W2, const float* __restrict__ b2,
                            const float* __restrict__ g2, const float* __restrict__ be2,
                            const float* __restrict__ W3, const float* __restrict__ b3,
                            float* __restrict__ out) {
    __shared__ float x1[1024], x2[512], x3[256], z[3];
    int b = blockIdx.x, tid = threadIdx.x;
    for (int i = tid; i < 1024; i += blockDim.x) x1[i] = l3[(size_t)b * 1024 + i];
    __syncthreads();
    for (int o = tid; o < 512; o += blockDim.x) {
        float acc = 0.f;
        for (int k = 0; k < 1024; ++k) acc += x1[k] * W1[(size_t)k * 512 + o];
        x2[o] = fmaxf((acc + b1[o]) * BNS * g1[o] + be1[o], 0.f);
    }
    __syncthreads();
    for (int o = tid; o < 256; o += blockDim.x) {
        float acc = 0.f;
        for (int k = 0; k < 512; ++k) acc += x2[k] * W2[(size_t)k * 256 + o];
        x3[o] = fmaxf((acc + b2[o]) * BNS * g2[o] + be2[o], 0.f);
    }
    __syncthreads();
    if (tid < 3) {
        float acc = 0.f;
        for (int k = 0; k < 256; ++k) acc += x3[k] * W3[(size_t)k * 3 + tid];
        z[tid] = acc + b3[tid];
    }
    __syncthreads();
    if (tid < 3) {
        float m = fmaxf(z[0], fmaxf(z[1], z[2]));
        float s = expf(z[0] - m) + expf(z[1] - m) + expf(z[2] - m);
        out[(size_t)b * 3 + tid] = z[tid] - m - logf(s);
    }
}

// ---------------------------------------------------------------- host
extern "C" void kernel_launch(void* const* d_in, const int* in_sizes, int n_in,
                              void* d_out, int out_size, void* d_ws, size_t ws_size,
                              hipStream_t stream) {
    (void)in_sizes; (void)n_in; (void)out_size; (void)ws_size;
    const int B = 8, N = 4096;
    char* ws = (char*)d_ws;
    size_t cur = 0;
    auto alloc = [&](size_t bytes) -> char* {
        char* p = ws + cur;
        cur = (cur + bytes + 255) & ~(size_t)255;
        return p;
    };
    float* xyz     = (float*)alloc((size_t)B * N * 2 * 4);
    float* l1_xyz  = (float*)alloc((size_t)B * 512 * 2 * 4);
    float* l2_xyz  = (float*)alloc((size_t)B * 128 * 2 * 4);
    int*   fps1    = (int*)alloc((size_t)B * 512 * 4);
    int*   fps2    = (int*)alloc((size_t)B * 128 * 4);
    int*   ballidx = (int*)alloc((size_t)B * 512 * 128 * 4);
    _Float16* l1_pts = (_Float16*)alloc((size_t)B * 512 * 320 * 2);
    _Float16* l2_pts = (_Float16*)alloc((size_t)B * 128 * 640 * 2);

    // layer table: {d_in index of W, K, C}; b,gamma,beta at idx+1..+3
    static const int LI[21] = {1,5,9, 13,17,21, 25,29,33,
                               37,41,45, 49,53,57, 61,65,69, 73,77,81};
    static const int LK[21] = {2,32,32, 2,64,64, 2,64,96,
                               322,64,64, 322,128,128, 322,128,128, 642,256,512};
    static const int LC[21] = {32,32,64, 64,64,128, 64,96,128,
                               64,64,128, 128,128,256, 128,128,256, 256,512,1024};
    _Float16* wf[21]; float* sb[21];
    for (int i = 0; i < 21; ++i) {
        int K = LK[i], C = LC[i], Kpad = (K + 31) & ~31;
        wf[i] = (_Float16*)alloc((size_t)Kpad * C * 2);
        sb[i] = (float*)alloc((size_t)2 * C * 4);
        int total = Kpad * C;
        prep_weights<<<(total + 255) / 256, 256, 0, stream>>>(
            (const float*)d_in[LI[i]], (const float*)d_in[LI[i] + 1],
            (const float*)d_in[LI[i] + 2], (const float*)d_in[LI[i] + 3],
            K, Kpad, C, wf[i], sb[i]);
    }

    transpose_xy<<<(B * N + 255) / 256, 256, 0, stream>>>((const float*)d_in[0], xyz, B, N);
    fps_kernel<<<B, 256, 0, stream>>>(xyz, N, 512, fps1, l1_xyz);

    float* l3 = (float*)d_out + 24;   // l3_pts region of output (also head input)

    // ---------------- SA1 : src=xyz(4096), centers=l1_xyz(512), out l1_pts stride 320
    ballquery_kernel<<<(B * 512) / 4, 128, 0, stream>>>(xyz, l1_xyz, N, 512, 0.01f, 16, ballidx);
    sa_fused<0, 32, 0, 32, 32, 64, 16, 4, false>
        <<<(B * 512) / 4, 128, 4 * 16 * (32 + 32) * 2 + 4 * 64 * 4, stream>>>(
            xyz, nullptr, l1_xyz, ballidx, wf[0], wf[1], wf[2], sb[0], sb[1], sb[2],
            l1_pts, 512, N, 320, 0);
    ballquery_kernel<<<(B * 512) / 4, 128, 0, stream>>>(xyz, l1_xyz, N, 512, 0.04f, 32, ballidx);
    sa_fused<0, 32, 0, 64, 64, 128, 32, 2, false>
        <<<(B * 512) / 2, 128, 4 * 16 * (64 + 64) * 2 + 2 * 128 * 4, stream>>>(
            xyz, nullptr, l1_xyz, ballidx, wf[3], wf[4], wf[5], sb[3], sb[4], sb[5],
            l1_pts, 512, N, 320, 64);
    ballquery_kernel<<<(B * 512) / 4, 128, 0, stream>>>(xyz, l1_xyz, N, 512, 0.16f, 128, ballidx);
    sa_fused<0, 32, 0, 64, 96, 128, 128, 1, false>
        <<<(B * 512), 128, 4 * 16 * (96 + 96) * 2 + 128 * 4, stream>>>(
            xyz, nullptr, l1_xyz, ballidx, wf[6], wf[7], wf[8], sb[6], sb[7], sb[8],
            l1_pts, 512, N, 320, 192);

    // ---------------- SA2 : src=l1 (512 pts, 320ch), centers=l2_xyz(128), out l2_pts stride 640
    fps_kernel<<<B, 256, 0, stream>>>(l1_xyz, 512, 128, fps2, l2_xyz);
    ballquery_kernel<<<(B * 128) / 4, 128, 0, stream>>>(l1_xyz, l2_xyz, 512, 128, 0.04f, 32, ballidx);
    sa_fused<1, 352, 320, 64, 64, 128, 32, 2, false>
        <<<(B * 128) / 2, 128, 4 * 16 * (352 + 64) * 2 + 2 * 128 * 4, stream>>>(
            l1_xyz, l1_pts, l2_xyz, ballidx, wf[9], wf[10], wf[11], sb[9], sb[10], sb[11],
            l2_pts, 128, 512, 640, 0);
    ballquery_kernel<<<(B * 128) / 4, 128, 0, stream>>>(l1_xyz, l2_xyz, 512, 128, 0.16f, 64, ballidx);
    sa_fused<1, 352, 320, 128, 128, 256, 64, 1, false>
        <<<(B * 128), 128, 4 * 16 * (352 + 128) * 2 + 256 * 4, stream>>>(
            l1_xyz, l1_pts, l2_xyz, ballidx, wf[12], wf[13], wf[14], sb[12], sb[13], sb[14],
            l2_pts, 128, 512, 640, 128);
    ballquery_kernel<<<(B * 128) / 4, 128, 0, stream>>>(l1_xyz, l2_xyz, 512, 128, 0.64f, 128, ballidx);
    sa_fused<1, 352, 320, 128, 128, 256, 128, 1, false>
        <<<(B * 128), 128, 4 * 16 * (352 + 128) * 2 + 256 * 4, stream>>>(
            l1_xyz, l1_pts, l2_xyz, ballidx, wf[15], wf[16], wf[17], sb[15], sb[16], sb[17],
            l2_pts, 128, 512, 640, 384);

    // ---------------- SA3 : global max over 128 pts, g=[xyz, pts], out l3 (f32)
    sa_fused<2, 672, 640, 256, 512, 1024, 128, 1, true>
        <<<B, 32, 1 * 16 * (672 + 512) * 2 + 1024 * 4, stream>>>(
            l2_xyz, l2_pts, nullptr, nullptr, wf[18], wf[19], wf[20], sb[18], sb[19], sb[20],
            l3, 1, 128, 1024, 0);

    // ---------------- head
    head_kernel<<<B, 256, 0, stream>>>(
        l3,
        (const float*)d_in[85], (const float*)d_in[86], (const float*)d_in[87], (const float*)d_in[88],
        (const float*)d_in[89], (const float*)d_in[90], (const float*)d_in[91], (const float*)d_in[92],
        (const float*)d_in[93], (const float*)d_in[94],
        (float*)d_out);
    (void)fps1; (void)fps2;
}
